// NPInstantaneousTransitionPrior_36610301231420
// MI455X (gfx1250) — matrix operand order repaired and checked by reading, not compiled, not run on hardware
//
#include <hip/hip_runtime.h>
#include <hip/hip_bf16.h>
#include <math.h>

// ---------------------------------------------------------------------------
// NPInstantaneousTransitionPrior on MI455X (gfx1250)
// - all layer GEMMs on v_wmma_f32_16x16x32_f16 (f16 in, f32 accumulate)
// - weights pre-converted once to f16 fragment layouts in d_ws
// - weight staging into LDS via Tensor Data Mover (tensor_load_to_lds),
//   double-buffered -> only 4 workgroup barriers per i-iteration
// - leaky-relu derivative recovered from activation sign bits (no d storage)
// One wave = 16 rows, 4 waves/block = 64 rows, 2040 blocks (exact).
// ---------------------------------------------------------------------------

typedef _Float16 half8 __attribute__((ext_vector_type(8)));
typedef _Float16 v16h  __attribute__((ext_vector_type(16)));
typedef float    v8f   __attribute__((ext_vector_type(8)));
typedef unsigned int v4u __attribute__((ext_vector_type(4)));
typedef int      v4i   __attribute__((ext_vector_type(4)));
typedef int      v8i   __attribute__((ext_vector_type(8)));

#define LRELU_SLOPE 0.2f

constexpr int BSZ  = 256;
constexpr int TLEN = 512;
constexpr int DIN  = 8;
constexpr int LW   = 2;
constexpr int H    = 128;
constexpr int NMID = 2;
constexpr int NWIN  = TLEN - LW;            // 510
constexpr int NROWS = BSZ * NWIN;           // 130560
constexpr int MAXIN = LW * DIN + DIN;       // 24

constexpr int ROWS_PER_WAVE  = 16;
constexpr int WAVES          = 4;
constexpr int NTHREADS       = WAVES * 32;              // 128
constexpr int ROWS_PER_BLOCK = ROWS_PER_WAVE * WAVES;   // 64
constexpr int NBLOCKS        = NROWS / ROWS_PER_BLOCK;  // 2040 (exact)

constexpr size_t RES_OFF    = 0;
constexpr size_t LOGDET_OFF = (size_t)NROWS * DIN;      // 1,044,480
constexpr size_t HJ_OFF     = LOGDET_OFF + BSZ;         // 1,044,736

// ---------------- workspace (d_ws) layout: pre-converted f16 weights -------
constexpr size_t WS_BF  = 0;        // [i][m][n][k] = Wmid[i,m,k,n] (fwd B)
constexpr size_t WS_BB  = 262144;   // [i][m][n][k] = Wmid[i,m,n,k] (bwd B)
constexpr size_t WS_W0C = 524288;   // [i][n][k<32] = W0[i,k,n] (padded)
constexpr size_t WS_W0R = 557056;   // [i][n<32][k] = W0[i,n,k] (padded)

// --------------------------- dynamic LDS layout ----------------------------
constexpr int OFF_WMA  = 0;        // f16[128*128] weight buffer A
constexpr int OFF_WMB  = 32768;    // f16[128*128] weight buffer B
constexpr int OFF_W0C  = 65536;    // f16[128*32]
constexpr int OFF_W0R  = 73728;    // f16[32*128]
constexpr int OFF_BIAS = 81920;    // float[3*128]: b0, bmid0, bmid1
constexpr int OFF_WOUT = 83456;    // float[128]
constexpr int OFF_PW   = 83968;    // per-wave regions
constexpr int PW_SIZE  = 12288;
constexpr int PW_X     = 0;        // f16[16*128]: inp (16x32) -> a2 -> u chain
constexpr int PW_Y     = 4096;     // f16[16*128]: a0
constexpr int PW_Z     = 8192;     // f16[16*128]: a1
constexpr size_t SMEM_BYTES = OFF_PW + (size_t)WAVES * PW_SIZE;  // 133,120 B

// ----------------------- WMMA fragment loads (LDS) -------------------------
__device__ __forceinline__ v16h load_a_frag(const _Float16* buf, int rowStride,
                                            int kbase, int lane) {
  const int row  = lane & 15;
  const int koff = (lane >> 4) * 8;
  const half8* p = (const half8*)(buf + row * rowStride + kbase + koff);
  half8 lo = p[0];        // K +0..7
  half8 hi = p[2];        // K +16..23
  return __builtin_shufflevector(lo, hi, 0, 1, 2, 3, 4, 5, 6, 7,
                                 8, 9, 10, 11, 12, 13, 14, 15);
}

__device__ __forceinline__ v16h load_b_frag(const _Float16* buf, int kStride,
                                            int ntile, int kbase, int lane) {
  const int n  = ntile * 16 + (lane & 15);
  const int ks = kbase + (lane >> 4) * 8;
  const half8* p = (const half8*)(buf + n * kStride + ks);
  return __builtin_shufflevector(p[0], p[2], 0, 1, 2, 3, 4, 5, 6, 7,
                                 8, 9, 10, 11, 12, 13, 14, 15);
}

// ------------------- TDM copy: linear f16 global -> LDS --------------------
#if defined(__has_builtin) && __has_builtin(__builtin_amdgcn_tensor_load_to_lds)
#define USE_TDM 1
#else
#define USE_TDM 0
#endif

#if USE_TDM
// n8 = number of 8-byte elements (multiple of 16B-aligned payload)
__device__ __forceinline__ void tdm_copy(void* ldsDst, const void* gSrc,
                                         unsigned n8) {
  // flat LDS pointer: low 32 bits are the DS byte address (ISA 10.2)
  unsigned laddr = (unsigned)(unsigned long long)(uintptr_t)ldsDst;
  unsigned long long ga = (unsigned long long)(uintptr_t)gSrc;
  v4u g0;
  g0[0] = 1u;                                            // count=1, user mode
  g0[1] = laddr;                                         // lds_addr
  g0[2] = (unsigned)(ga & 0xFFFFFFFFu);                  // global_addr[31:0]
  g0[3] = (unsigned)((ga >> 32) & 0x1FFFFFFu) | (2u << 30);  // [56:32] | type=2
  v8i g1;
  g1[0] = (3 << 16);                       // data_size=3 (8B), mask=0
  g1[1] = (int)((n8 & 0xFFFFu) << 16);     // tensor_dim0[15:0] -> bits 63:48
  g1[2] = (int)((n8 >> 16) & 0xFFFFu) | (1 << 16);  // tdim0 hi | tensor_dim1=1
  g1[3] = (int)((n8 & 0xFFFFu) << 16);     // tile_dim0 -> bits 127:112
  g1[4] = 1;                               // tile_dim1=1, tile_dim2=0
  g1[5] = (int)n8;                         // tensor_dim0_stride lo32
  g1[6] = 0;
  g1[7] = 0;
  v4i gz = {0, 0, 0, 0};
#if __clang_major__ >= 23
  v8i gz8 = {0, 0, 0, 0, 0, 0, 0, 0};
  __builtin_amdgcn_tensor_load_to_lds(g0, g1, gz, gz, gz8, 0);
#else
  __builtin_amdgcn_tensor_load_to_lds(g0, g1, gz, gz, 0);
#endif
}
#else
__device__ __forceinline__ void copy_lds16(_Float16* dst, const _Float16* src,
                                           int halfs, int tid) {
  const half8* s = (const half8*)src;
  half8* d = (half8*)dst;
  for (int idx = tid; idx < (halfs >> 3); idx += NTHREADS) d[idx] = s[idx];
}
#endif

// ------------------- one GEMM layer tile (16 x 128 x 32*KT) ----------------
// FWD=true : C init = bias, leaky-relu, f16 out
// FWD=false: C init = 0, multiply by d recovered from sign(signSrc), f16 out
template <int KT, int INSTR, bool FWD>
__device__ __forceinline__ void wmma_tile(const _Float16* ain, _Float16* aout,
                                          const _Float16* wbuf,
                                          const float* bias,
                                          const _Float16* signSrc, int lane) {
  v16h afr[KT];
#pragma unroll
  for (int kt = 0; kt < KT; ++kt)
    afr[kt] = load_a_frag(ain, INSTR, kt * 32, lane);

  const int col   = lane & 15;
  const int rbase = (lane >> 4) * 8;
  const unsigned short* sgn = (const unsigned short*)signSrc;

#pragma unroll
  for (int nt = 0; nt < 8; ++nt) {
    v8f c;
    float bv = 0.0f;
    if constexpr (FWD) bv = bias[nt * 16 + col];
#pragma unroll
    for (int r = 0; r < 8; ++r) c[r] = bv;

#pragma unroll
    for (int kt = 0; kt < KT; ++kt) {
      v16h bfr = load_b_frag(wbuf, H, nt, kt * 32, lane);
      c = __builtin_amdgcn_wmma_f32_16x16x32_f16(false, afr[kt], false, bfr,
                                                 (short)0, c, false, false);
    }

#pragma unroll
    for (int r = 0; r < 8; ++r) {
      const int off = (r + rbase) * H + nt * 16 + col;
      if constexpr (FWD) {
        float h = c[r];
        aout[off] = (_Float16)((h >= 0.0f) ? h : h * LRELU_SLOPE);
      } else {
        float d = (sgn[off] & 0x8000u) ? LRELU_SLOPE : 1.0f;
        aout[off] = (_Float16)(c[r] * d);
      }
    }
  }
}

// ------------------------------ pre-pass -----------------------------------
__global__ void prep_kernel(const float* __restrict__ W0,
                            const float* __restrict__ Wmid,
                            _Float16* __restrict__ ws,
                            float* __restrict__ out) {
  const int tid    = blockIdx.x * blockDim.x + threadIdx.x;
  const int stride = gridDim.x * blockDim.x;

  for (int idx = tid; idx < BSZ; idx += stride) out[LOGDET_OFF + idx] = 0.0f;

  for (int idx = tid; idx < DIN * NMID * H * H; idx += stride)
    ws[WS_BB + idx] = (_Float16)Wmid[idx];

  for (int idx = tid; idx < DIN * NMID * H * H; idx += stride) {
    int k = idx & 127, n = (idx >> 7) & 127, im = idx >> 14;
    ws[WS_BF + idx] = (_Float16)Wmid[((size_t)im * H + k) * H + n];
  }

  for (int idx = tid; idx < DIN * H * 32; idx += stride) {
    int k = idx & 31, n = (idx >> 5) & 127, i = idx >> 12;
    float v = (k < 17 + i) ? W0[(size_t)(i * MAXIN + k) * H + n] : 0.0f;
    ws[WS_W0C + idx] = (_Float16)v;
  }

  for (int idx = tid; idx < DIN * 32 * H; idx += stride) {
    int k = idx & 127, n = (idx >> 7) & 31, i = idx >> 12;
    float v = (n < 17 + i) ? W0[(size_t)(i * MAXIN + n) * H + k] : 0.0f;
    ws[WS_W0R + idx] = (_Float16)v;
  }
}

// ------------------------------ main kernel --------------------------------
__global__ void __launch_bounds__(NTHREADS, 2)
prior_kernel(const float* __restrict__ x, const float* __restrict__ alphas,
             const float* __restrict__ b0, const float* __restrict__ bmid,
             const float* __restrict__ Wout, const float* __restrict__ bout,
             const _Float16* __restrict__ ws, float* __restrict__ out) {
  extern __shared__ __align__(16) char smem[];
  _Float16* sWmA  = (_Float16*)(smem + OFF_WMA);
  _Float16* sWmB  = (_Float16*)(smem + OFF_WMB);
  _Float16* sW0c  = (_Float16*)(smem + OFF_W0C);
  _Float16* sW0r  = (_Float16*)(smem + OFF_W0R);
  float*    sBias = (float*)(smem + OFF_BIAS);   // [0]=b0 [128]=bm0 [256]=bm1
  float*    sWout = (float*)(smem + OFF_WOUT);

  const int tid  = threadIdx.x;
  const int lane = tid & 31;
  const int w    = tid >> 5;
  char* pw = smem + OFF_PW + (size_t)w * PW_SIZE;
  _Float16* sX = (_Float16*)(pw + PW_X);  // inp -> a2 -> u2/u1/u0
  _Float16* sY = (_Float16*)(pw + PW_Y);  // a0
  _Float16* sZ = (_Float16*)(pw + PW_Z);  // a1

  const int rowBlock = blockIdx.x * ROWS_PER_BLOCK + w * ROWS_PER_WAVE;

  for (int i = 0; i < DIN; ++i) {
    __syncthreads();  // B0: prior i's readers of all shared buffers are done

    // ---- stage all forward weights (TDM or vector copies) ----
#if USE_TDM
    if (w == 0) {
      tdm_copy(sW0c, ws + WS_W0C + (size_t)i * 4096, 1024);
      tdm_copy(sW0r, ws + WS_W0R + (size_t)i * 4096, 1024);
      tdm_copy(sWmA, ws + WS_BF + (size_t)(i * NMID + 0) * 16384, 4096);
      tdm_copy(sWmB, ws + WS_BF + (size_t)(i * NMID + 1) * 16384, 4096);
    }
#else
    copy_lds16(sW0c, ws + WS_W0C + (size_t)i * 4096, 4096, tid);
    copy_lds16(sW0r, ws + WS_W0R + (size_t)i * 4096, 4096, tid);
    copy_lds16(sWmA, ws + WS_BF + (size_t)(i * NMID + 0) * 16384, 16384, tid);
    copy_lds16(sWmB, ws + WS_BF + (size_t)(i * NMID + 1) * 16384, 16384, tid);
#endif
    // biases + Wout (tid < 128 == all threads)
    sBias[tid]       = b0[i * H + tid];
    sBias[128 + tid] = bmid[(i * NMID + 0) * H + tid];
    sBias[256 + tid] = bmid[(i * NMID + 1) * H + tid];
    sWout[tid]       = Wout[i * H + tid];

    // ---- build padded 16x32 input tile into X (wave-local) ----
    if (lane < 16) {
      int rg = rowBlock + lane;
      int b  = rg / NWIN;
      int t  = rg - b * NWIN;
      const float* xp = x + (size_t)(b * TLEN + t) * DIN;  // 24 contiguous f32
      _Float16* ip = sX + lane * 32;
#pragma unroll
      for (int j = 0; j < 2 * DIN; ++j) ip[j] = (_Float16)xp[j];
      for (int j = 0; j < i; ++j)
        ip[16 + j] = (_Float16)(xp[16 + j] * alphas[i * DIN + j]);
      ip[16 + i] = (_Float16)xp[16 + i];
      for (int j = 17 + i; j < 32; ++j) ip[j] = (_Float16)0.0f;
    }
#if USE_TDM
    if (w == 0) __builtin_amdgcn_s_wait_tensorcnt((short)0);
#endif
    __syncthreads();  // B1: weights + biases visible

    // ---- forward chain, no barriers (activations are wave-private) ----
    wmma_tile<1, 32, true>(sX, sY, sW0c, sBias, nullptr, lane);        // a0
    wmma_tile<4, H,  true>(sY, sZ, sWmA, sBias + 128, nullptr, lane);  // a1
    wmma_tile<4, H,  true>(sZ, sX, sWmB, sBias + 256, nullptr, lane);  // a2

    // residual: res = a2 @ Wout + bout (half-row split + xor-shuffle)
    {
      int row = lane & 15, hh = lane >> 4;
      const _Float16* ar = sX + row * H + hh * 64;
      const float*    wo = sWout + hh * 64;
      float s = 0.0f;
#pragma unroll
      for (int h = 0; h < 64; ++h) s += (float)ar[h] * wo[h];
      s += __shfl_xor(s, 16);
      if (lane < 16)
        out[RES_OFF + (size_t)(rowBlock + row) * DIN + i] = s + bout[i];
    }
    // u2 = Wout * d2 (d2 = sign of a2), in place in X
    for (int idx = lane; idx < 16 * H; idx += 32) {
      unsigned short s2 = ((const unsigned short*)sX)[idx];
      float d = (s2 & 0x8000u) ? LRELU_SLOPE : 1.0f;
      sX[idx] = (_Float16)(sWout[idx & 127] * d);
    }
    __syncthreads();  // B2: all waves done reading WmA/WmB (forward)

    // ---- stage backward weights ----
#if USE_TDM
    if (w == 0) {
      tdm_copy(sWmA, ws + WS_BB + (size_t)(i * NMID + 1) * 16384, 4096);
      tdm_copy(sWmB, ws + WS_BB + (size_t)(i * NMID + 0) * 16384, 4096);
      __builtin_amdgcn_s_wait_tensorcnt((short)0);
    }
#else
    copy_lds16(sWmA, ws + WS_BB + (size_t)(i * NMID + 1) * 16384, 16384, tid);
    copy_lds16(sWmB, ws + WS_BB + (size_t)(i * NMID + 0) * 16384, 16384, tid);
#endif
    __syncthreads();  // B3

    // ---- backward chain (in place in X; A fragments preloaded) ----
    wmma_tile<4, H, false>(sX, sX, sWmA, nullptr, sZ, lane);  // u1 (*d1)
    wmma_tile<4, H, false>(sX, sX, sWmB, nullptr, sY, lane);  // u0 (*d0)

    // ---- g = u0 @ W0^T (K=128, N=32 padded) -> direct global emit ----
    {
      v16h afr[4];
#pragma unroll
      for (int kt = 0; kt < 4; ++kt)
        afr[kt] = load_a_frag(sX, H, kt * 32, lane);
      const int col = lane & 15, rbase = (lane >> 4) * 8;

      v8f cg[2];
#pragma unroll
      for (int nt = 0; nt < 2; ++nt) {
#pragma unroll
        for (int r = 0; r < 8; ++r) cg[nt][r] = 0.0f;
#pragma unroll
        for (int kt = 0; kt < 4; ++kt) {
          v16h bfr = load_b_frag(sW0r, H, nt, kt * 32, lane);
          cg[nt] = __builtin_amdgcn_wmma_f32_16x16x32_f16(
              false, afr[kt], false, bfr, (short)0, cg[nt], false, false);
        }
      }

      const size_t hjBase =
          HJ_OFF + (size_t)NROWS * (size_t)(16 * i + (i * (i - 1)) / 2);
      const int ncols = 16 + i;
#pragma unroll
      for (int nt = 0; nt < 2; ++nt) {
        int j = nt * 16 + col;
#pragma unroll
        for (int r = 0; r < 8; ++r) {
          int rg = rowBlock + rbase + r;
          if (j < ncols) {
            out[hjBase + (size_t)rg * (size_t)ncols + j] = cg[nt][r];
          } else if (j == ncols) {  // last column -> logdet
            atomicAdd(out + LOGDET_OFF + rg / NWIN, logf(fabsf(cg[nt][r])));
          }
        }
      }
    }
  }
}

// --------------------------------- launch ----------------------------------
extern "C" void kernel_launch(void* const* d_in, const int* in_sizes, int n_in,
                              void* d_out, int out_size, void* d_ws,
                              size_t ws_size, hipStream_t stream) {
  const float* x      = (const float*)d_in[0];
  const float* alphas = (const float*)d_in[1];
  const float* W0     = (const float*)d_in[2];
  const float* b0     = (const float*)d_in[3];
  const float* Wmid   = (const float*)d_in[4];
  const float* bmid   = (const float*)d_in[5];
  const float* Wout   = (const float*)d_in[6];
  const float* bout   = (const float*)d_in[7];
  float* out = (float*)d_out;
  _Float16* ws = (_Float16*)d_ws;

  prep_kernel<<<256, 256, 0, stream>>>(W0, Wmid, ws, out);
  prior_kernel<<<NBLOCKS, NTHREADS, SMEM_BYTES, stream>>>(
      x, alphas, b0, bmid, Wout, bout, ws, out);
}
